// _TSTEncoderLayer_3874060501013
// MI455X (gfx1250) — matrix operand order repaired and checked by reading, not compiled, not run on hardware
//
#include <hip/hip_runtime.h>
#include <hip/hip_bf16.h>
#include <math.h>

typedef __bf16 bf16;
typedef __attribute__((ext_vector_type(16))) __bf16 v16bf;
typedef __attribute__((ext_vector_type(8)))  float   v8f;
typedef __attribute__((ext_vector_type(4)))  unsigned int u32x4;
typedef __attribute__((ext_vector_type(8)))  int     i32x8;
typedef __attribute__((ext_vector_type(4)))  int     i32x4;

#define D_MODEL 512
#define N_HEADS 8
#define D_HEAD  64
#define D_FF    2048
#define BATCH   16
#define SEQ     1024
#define ROWS    (BATCH * SEQ)   // 16384
#define BN_EPS  1e-5f

#if defined(__has_builtin)
#  if __has_builtin(__builtin_amdgcn_tensor_load_to_lds)
#    define TDM_OK 1
#  else
#    define TDM_OK 0
#  endif
#else
#  define TDM_OK 0
#endif

// ---------------------------------------------------------------------------
// WMMA fragment helpers (layouts per CDNA5 ISA 7.12.2, wave32)
// A 16x32 bf16: lane l=L&15 -> M=l ; element e -> K = (e/8)*16 + (L>>4)*8 + (e&7)
// B 32x16 bf16 (from [N x K] row-major storage): lane -> N=L&15, K = (e/8)*16 + (L>>4)*8 + (e&7)
// C/D 16x16 f32: vgpr r -> row = r + (L>>4)*8 ; col = L&15.
// ---------------------------------------------------------------------------
__device__ __forceinline__ v16bf load_frag_a(const bf16* base, int ld, int row0, int k0) {
  const int lane = threadIdx.x & 31;
  const int l = lane & 15, hi = lane >> 4;
  const bf16* p = base + (size_t)(row0 + l) * ld + k0 + hi * 8;
  union { u32x4 q[2]; v16bf v; } u;
  u.q[0] = *(const u32x4*)(p);
  u.q[1] = *(const u32x4*)(p + 16);
  return u.v;
}

__device__ __forceinline__ v16bf load_frag_b(const bf16* baseT, int ld, int col0, int k0) {
  const int lane = threadIdx.x & 31;
  const int n = col0 + (lane & 15), hi = lane >> 4;
  const bf16* p = baseT + (size_t)n * ld + k0 + hi * 8;
  union { u32x4 q[2]; v16bf v; } u;
  u.q[0] = *(const u32x4*)(p);
  u.q[1] = *(const u32x4*)(p + 16);
  return u.v;
}

__device__ __forceinline__ v8f wmma_bf16(v16bf a, v16bf b, v8f c) {
  return __builtin_amdgcn_wmma_f32_16x16x32_bf16(false, a, false, b, (short)0, c, false, false);
}

__device__ __forceinline__ float gelu_exact(float x) {
  return 0.5f * x * (1.0f + erff(x * 0.70710678118654752f));
}

// ---------------------------------------------------------------------------
// TDM: 2D tile load (bf16 elements) global -> LDS.  D# layout per ISA ch.8:
//  group0: [1:0]=count=1 | [63:32]=lds_addr | [120:64]=global_addr | [127:126]=2
//  group1: [17:16]=data_size(1=2B) | tensor_dim0[79:48] | tensor_dim1[111:80]
//          | tile_dim0[127:112] | tile_dim1[143:128] | dim0_stride[207:160]
// ---------------------------------------------------------------------------
__device__ __forceinline__ void tdm_load_tile(unsigned int lds_addr, const void* g,
                                              unsigned int tile_cols, unsigned int tile_rows,
                                              unsigned int row_stride_elems) {
#if TDM_OK
  const unsigned long long ga = (unsigned long long)(size_t)g;
  u32x4 g0 = {};
  g0.x = 1u;                                                  // count=1, user mode
  g0.y = lds_addr;
  g0.z = (unsigned int)ga;
  g0.w = ((unsigned int)(ga >> 32) & 0x01FFFFFFu) | (2u << 30); // addr[56:32] | type=2
  i32x8 g1 = {};
  g1[0] = (int)(1u << 16);                                    // data_size = 2 bytes
  g1[1] = (int)((row_stride_elems & 0xFFFFu) << 16);          // tensor_dim0 lo16
  g1[2] = (int)(((row_stride_elems >> 16) & 0xFFFFu) | ((tile_rows & 0xFFFFu) << 16));
  g1[3] = (int)(((tile_rows >> 16) & 0xFFFFu) | ((tile_cols & 0xFFFFu) << 16)); // tile_dim0
  g1[4] = (int)(tile_rows & 0xFFFFu);                         // tile_dim1
  g1[5] = (int)row_stride_elems;                              // dim0_stride lo32
  i32x4 zz = {};
#if defined(__clang_major__) && (__clang_major__ >= 23)
  i32x8 z8 = {};
  __builtin_amdgcn_tensor_load_to_lds(g0, g1, zz, zz, z8, 0);
#else
  __builtin_amdgcn_tensor_load_to_lds(g0, g1, zz, zz, 0);
#endif
#else
  (void)lds_addr; (void)g; (void)tile_cols; (void)tile_rows; (void)row_stride_elems;
#endif
}

// ---------------------------------------------------------------------------
// bf16 WMMA GEMM: C[M,N] = A[M,K] * Bt[N,K]^T.  Block = 256 thr = 8 waves
// (4x2), wave tile 32x32, block tile 128x64.  TDM double-buffers the
// 128x32 A tile and 64x32 B tile in LDS.
// ---------------------------------------------------------------------------
enum { EPI_BF16 = 0, EPI_VT = 1, EPI_RESF32 = 2, EPI_GELU = 3 };

template<int EPI>
__device__ __forceinline__ void store_tile(v8f c, int row0, int col0, int N,
                                           const float* bias, const float* resid,
                                           float* outF, bf16* outB, float scale) {
  const int lane = threadIdx.x & 31;
  const int l = lane & 15, hi = lane >> 4;
  const int col = col0 + l;
  const float bv = bias ? bias[col] : 0.0f;
#pragma unroll
  for (int r = 0; r < 8; ++r) {
    const int row = row0 + r + hi * 8;
    float v = c[r] * scale + bv;
    const size_t idx = (size_t)row * N + col;
    if (EPI == EPI_BF16) {
      outB[idx] = (bf16)v;
    } else if (EPI == EPI_VT) {              // V projection -> [b,h,d,t] layout
      const int b = row >> 10, t = row & 1023;
      const int h = col >> 6,  d = col & 63;
      outB[(((size_t)(b * N_HEADS + h) * D_HEAD + d) << 10) + t] = (bf16)v;
    } else if (EPI == EPI_RESF32) {          // residual add, fp32 out
      outF[idx] = v + resid[idx];
    } else {                                 // EPI_GELU
      outB[idx] = (bf16)gelu_exact(v);
    }
  }
}

template<int EPI>
__global__ __launch_bounds__(256)
void gemm_wmma(const bf16* __restrict__ A, const bf16* __restrict__ Bt,
               const float* __restrict__ bias, const float* __restrict__ resid,
               float* __restrict__ outF, bf16* __restrict__ outB,
               int M, int N, int K, float scale) {
  __shared__ __align__(16) bf16 sA[2][128 * 32];
  __shared__ __align__(16) bf16 sB[2][64 * 32];
  const int wave = threadIdx.x >> 5;
  const int row0 = blockIdx.y * 128;
  const int col0 = blockIdx.x * 64;
  const int wr = (wave >> 1) * 32;
  const int wc = (wave & 1) * 32;
  const int nk = K >> 5;
  v8f c00 = {}, c01 = {}, c10 = {}, c11 = {};

#if TDM_OK
  const unsigned int aAddr[2] = {(unsigned int)(size_t)&sA[0][0], (unsigned int)(size_t)&sA[1][0]};
  const unsigned int bAddr[2] = {(unsigned int)(size_t)&sB[0][0], (unsigned int)(size_t)&sB[1][0]};
  if (wave == 0) {
    tdm_load_tile(aAddr[0], A + (size_t)row0 * K, 32, 128, K);
    tdm_load_tile(bAddr[0], Bt + (size_t)col0 * K, 32, 64, K);
  }
  for (int i = 0; i < nk; ++i) {
    const int cur = i & 1, nxt = cur ^ 1;
    if (i + 1 < nk) {
      if (wave == 0) {
        tdm_load_tile(aAddr[nxt], A + (size_t)row0 * K + (size_t)(i + 1) * 32, 32, 128, K);
        tdm_load_tile(bAddr[nxt], Bt + (size_t)col0 * K + (size_t)(i + 1) * 32, 32, 64, K);
      }
      __builtin_amdgcn_s_wait_tensorcnt(2);   // current pair complete (in-order)
    } else {
      __builtin_amdgcn_s_wait_tensorcnt(0);
    }
    __syncthreads();                          // publish LDS tiles to all waves
    v16bf a0 = load_frag_a(&sA[cur][0], 32, wr,      0);
    v16bf a1 = load_frag_a(&sA[cur][0], 32, wr + 16, 0);
    v16bf b0 = load_frag_b(&sB[cur][0], 32, wc,      0);
    v16bf b1 = load_frag_b(&sB[cur][0], 32, wc + 16, 0);
    c00 = wmma_bf16(a0, b0, c00);
    c01 = wmma_bf16(a0, b1, c01);
    c10 = wmma_bf16(a1, b0, c10);
    c11 = wmma_bf16(a1, b1, c11);
    __syncthreads();                          // all waves done with buf[cur]
  }
#else
  for (int i = 0; i < nk; ++i) {
    __syncthreads();
    for (int v = threadIdx.x; v < 512; v += 256) {   // A tile: 512 x 16B
      const int r = v >> 2, ch = v & 3;
      ((u32x4*)&sA[0][0])[v] = *(const u32x4*)(A + (size_t)(row0 + r) * K + i * 32 + ch * 8);
    }
    if (threadIdx.x < 256) {                         // B tile: 256 x 16B
      const int v = threadIdx.x, r = v >> 2, ch = v & 3;
      ((u32x4*)&sB[0][0])[v] = *(const u32x4*)(Bt + (size_t)(col0 + r) * K + i * 32 + ch * 8);
    }
    __syncthreads();
    v16bf a0 = load_frag_a(&sA[0][0], 32, wr,      0);
    v16bf a1 = load_frag_a(&sA[0][0], 32, wr + 16, 0);
    v16bf b0 = load_frag_b(&sB[0][0], 32, wc,      0);
    v16bf b1 = load_frag_b(&sB[0][0], 32, wc + 16, 0);
    c00 = wmma_bf16(a0, b0, c00);
    c01 = wmma_bf16(a0, b1, c01);
    c10 = wmma_bf16(a1, b0, c10);
    c11 = wmma_bf16(a1, b1, c11);
  }
#endif

  store_tile<EPI>(c00, row0 + wr,      col0 + wc,      N, bias, resid, outF, outB, scale);
  store_tile<EPI>(c01, row0 + wr,      col0 + wc + 16, N, bias, resid, outF, outB, scale);
  store_tile<EPI>(c10, row0 + wr + 16, col0 + wc,      N, bias, resid, outF, outB, scale);
  store_tile<EPI>(c11, row0 + wr + 16, col0 + wc + 16, N, bias, resid, outF, outB, scale);
}

// ---------------------------------------------------------------------------
// Flash attention: grid (S/64, B*H). Block = 128 thr = 4 waves, each wave owns
// 16 query rows. K (32x64) and V^T (64x32) tiles are shared by all 4 waves ->
// TDM double-buffered in LDS.  Q pre-scaled by 1/sqrt(d_k).
// ---------------------------------------------------------------------------
__global__ __launch_bounds__(128)
void flash_attn(const bf16* __restrict__ Q, const bf16* __restrict__ Km,
                const bf16* __restrict__ Vt, bf16* __restrict__ ctx) {
  const int qt = blockIdx.x;
  const int bh = blockIdx.y;
  const int b = bh >> 3, h = bh & 7;
  const int wave = threadIdx.x >> 5;
  const int lane = threadIdx.x & 31;
  const int l = lane & 15, hi = lane >> 4;
  const int q0 = qt * 64 + wave * 16;
  const int NT = SEQ / 32;

  __shared__ __align__(16) bf16 pls[4][16][32];

  const bf16* Qb = Q  + (size_t)(b * SEQ) * D_MODEL + h * D_HEAD;
  const bf16* Kb = Km + (size_t)(b * SEQ) * D_MODEL + h * D_HEAD;
  const bf16* Vb = Vt + (size_t)bh * D_HEAD * SEQ;

  const v16bf qa0 = load_frag_a(Qb, D_MODEL, q0, 0);
  const v16bf qa1 = load_frag_a(Qb, D_MODEL, q0, 32);

  v8f o0 = {}, o1 = {}, o2 = {}, o3 = {};
  float mrow[8], lrow[8];
#pragma unroll
  for (int r = 0; r < 8; ++r) { mrow[r] = -1e30f; lrow[r] = 0.0f; }

#if TDM_OK
  __shared__ __align__(16) bf16 sK[2][32 * 64];   // [t][d]
  __shared__ __align__(16) bf16 sV[2][64 * 32];   // [d][t]
  const unsigned int kAddr[2] = {(unsigned int)(size_t)&sK[0][0], (unsigned int)(size_t)&sK[1][0]};
  const unsigned int vAddr[2] = {(unsigned int)(size_t)&sV[0][0], (unsigned int)(size_t)&sV[1][0]};
  if (wave == 0) {
    tdm_load_tile(kAddr[0], Kb, 64, 32, D_MODEL);
    tdm_load_tile(vAddr[0], Vb, 32, 64, SEQ);
  }
#endif

  for (int i = 0; i < NT; ++i) {
    const int t0 = i * 32;
    v8f s0 = {}, s1 = {};
#if TDM_OK
    const int cur = i & 1, nxt = cur ^ 1;
    if (i + 1 < NT) {
      if (wave == 0) {
        tdm_load_tile(kAddr[nxt], Kb + (size_t)(t0 + 32) * D_MODEL, 64, 32, D_MODEL);
        tdm_load_tile(vAddr[nxt], Vb + (t0 + 32), 32, 64, SEQ);
      }
      __builtin_amdgcn_s_wait_tensorcnt(2);
    } else {
      __builtin_amdgcn_s_wait_tensorcnt(0);
    }
    __syncthreads();
    const bf16* Kt = &sK[cur][0];
    const bf16* Vw = &sV[cur][0];
    s0 = wmma_bf16(qa0, load_frag_b(Kt, 64, 0,  0),  s0);
    s0 = wmma_bf16(qa1, load_frag_b(Kt, 64, 0,  32), s0);
    s1 = wmma_bf16(qa0, load_frag_b(Kt, 64, 16, 0),  s1);
    s1 = wmma_bf16(qa1, load_frag_b(Kt, 64, 16, 32), s1);
#else
    s0 = wmma_bf16(qa0, load_frag_b(Kb, D_MODEL, t0,      0),  s0);
    s0 = wmma_bf16(qa1, load_frag_b(Kb, D_MODEL, t0,      32), s0);
    s1 = wmma_bf16(qa0, load_frag_b(Kb, D_MODEL, t0 + 16, 0),  s1);
    s1 = wmma_bf16(qa1, load_frag_b(Kb, D_MODEL, t0 + 16, 32), s1);
#endif

    float alpha[8];
#pragma unroll
    for (int r = 0; r < 8; ++r) {
      float v = fmaxf(s0[r], s1[r]);
      for (int m = 1; m < 16; m <<= 1) v = fmaxf(v, __shfl_xor(v, m, 32));
      const float nm = fmaxf(mrow[r], v);
      alpha[r] = __expf(mrow[r] - nm);
      mrow[r] = nm;
      const float p0 = __expf(s0[r] - nm);
      const float p1 = __expf(s1[r] - nm);
      s0[r] = p0; s1[r] = p1;
      float rs = p0 + p1;
      for (int m = 1; m < 16; m <<= 1) rs += __shfl_xor(rs, m, 32);
      lrow[r] = lrow[r] * alpha[r] + rs;
    }
#pragma unroll
    for (int r = 0; r < 8; ++r) {             // rescale O; stash P for relayout
      o0[r] *= alpha[r]; o1[r] *= alpha[r]; o2[r] *= alpha[r]; o3[r] *= alpha[r];
      pls[wave][r + hi * 8][l]      = (bf16)s0[r];
      pls[wave][r + hi * 8][16 + l] = (bf16)s1[r];
    }
    // wave-private LDS tile; per-wave DS ordering makes the reload safe
    const v16bf pa = load_frag_a(&pls[wave][0][0], 32, 0, 0);
#if TDM_OK
    o0 = wmma_bf16(pa, load_frag_b(Vw, 32, 0,  0), o0);
    o1 = wmma_bf16(pa, load_frag_b(Vw, 32, 16, 0), o1);
    o2 = wmma_bf16(pa, load_frag_b(Vw, 32, 32, 0), o2);
    o3 = wmma_bf16(pa, load_frag_b(Vw, 32, 48, 0), o3);
    __syncthreads();                          // done with buf[cur] before reissue
#else
    o0 = wmma_bf16(pa, load_frag_b(Vb, SEQ, 0,  t0), o0);
    o1 = wmma_bf16(pa, load_frag_b(Vb, SEQ, 16, t0), o1);
    o2 = wmma_bf16(pa, load_frag_b(Vb, SEQ, 32, t0), o2);
    o3 = wmma_bf16(pa, load_frag_b(Vb, SEQ, 48, t0), o3);
#endif
  }

#pragma unroll
  for (int r = 0; r < 8; ++r) {
    const float inv = 1.0f / lrow[r];
    const size_t row  = (size_t)(b * SEQ + q0 + r + hi * 8);
    const size_t base = row * D_MODEL + h * D_HEAD + l;
    ctx[base]      = (bf16)(o0[r] * inv);
    ctx[base + 16] = (bf16)(o1[r] * inv);
    ctx[base + 32] = (bf16)(o2[r] * inv);
    ctx[base + 48] = (bf16)(o3[r] * inv);
  }
}

// ---------------------------------------------------------------------------
// BatchNorm helpers
// ---------------------------------------------------------------------------
__global__ __launch_bounds__(256)
void bn_stats(const float* __restrict__ x, float* __restrict__ sum, float* __restrict__ sumsq) {
  const int rowsPer = ROWS / gridDim.x;
  const int r0 = blockIdx.x * rowsPer;
  for (int c = threadIdx.x; c < D_MODEL; c += blockDim.x) {
    float s = 0.f, ss = 0.f;
    for (int r = 0; r < rowsPer; ++r) {
      const float v = x[(size_t)(r0 + r) * D_MODEL + c];
      s += v; ss += v * v;
    }
    atomicAdd(&sum[c], s);
    atomicAdd(&sumsq[c], ss);
  }
}

__global__ void bn_finalize(const float* __restrict__ sum, const float* __restrict__ sumsq,
                            const float* __restrict__ gamma, const float* __restrict__ beta,
                            float* __restrict__ scale, float* __restrict__ shift) {
  const int c = threadIdx.x;  // 512 threads, 1 block
  const float inv_n = 1.0f / (float)ROWS;
  const float mean = sum[c] * inv_n;
  const float var  = sumsq[c] * inv_n - mean * mean;
  const float sc   = gamma[c] * rsqrtf(var + BN_EPS);
  scale[c] = sc;
  shift[c] = beta[c] - mean * sc;
}

__global__ void bn_apply(const float* __restrict__ in, const float* __restrict__ scale,
                         const float* __restrict__ shift, float* __restrict__ outF,
                         bf16* __restrict__ outB, size_t n) {
  const size_t stride = (size_t)gridDim.x * blockDim.x;
  for (size_t i = (size_t)blockIdx.x * blockDim.x + threadIdx.x; i < n; i += stride) {
    const int c = (int)(i & (D_MODEL - 1));
    const float v = in[i] * scale[c] + shift[c];
    outF[i] = v;
    if (outB) outB[i] = (bf16)v;
  }
}

// ---------------------------------------------------------------------------
// fp32 -> bf16 conversions (plain / transposed-to-[NxK])
// ---------------------------------------------------------------------------
__global__ void cvt_bf16(const float* __restrict__ in, bf16* __restrict__ out, size_t n) {
  const size_t stride = (size_t)gridDim.x * blockDim.x;
  for (size_t i = (size_t)blockIdx.x * blockDim.x + threadIdx.x; i < n; i += stride)
    out[i] = (bf16)in[i];
}

__global__ void cvt_bf16_T(const float* __restrict__ in, bf16* __restrict__ out, int Kd, int Nd) {
  const size_t total = (size_t)Kd * Nd;
  const size_t stride = (size_t)gridDim.x * blockDim.x;
  for (size_t i = (size_t)blockIdx.x * blockDim.x + threadIdx.x; i < total; i += stride) {
    const size_t k = i / Nd, n = i % Nd;
    out[n * (size_t)Kd + k] = (bf16)in[i];
  }
}

// ---------------------------------------------------------------------------
extern "C" void kernel_launch(void* const* d_in, const int* in_sizes, int n_in,
                              void* d_out, int out_size, void* d_ws, size_t ws_size,
                              hipStream_t stream) {
  (void)in_sizes; (void)n_in; (void)out_size; (void)ws_size;
  const float* src    = (const float*)d_in[0];
  const float* W_Q    = (const float*)d_in[1];
  const float* W_K    = (const float*)d_in[2];
  const float* W_V    = (const float*)d_in[3];
  const float* W_O    = (const float*)d_in[4];
  const float* W1     = (const float*)d_in[5];
  const float* b1     = (const float*)d_in[6];
  const float* W2     = (const float*)d_in[7];
  const float* b2     = (const float*)d_in[8];
  const float* gamma1 = (const float*)d_in[9];
  const float* beta1  = (const float*)d_in[10];
  const float* gamma2 = (const float*)d_in[11];
  const float* beta2  = (const float*)d_in[12];

  const size_t nAct = (size_t)ROWS * D_MODEL;          // 8,388,608
  char* ws = (char*)d_ws;
  size_t o = 0;
  bf16* srcb = (bf16*)(ws + o); o += nAct * 2;
  bf16* WqT  = (bf16*)(ws + o); o += (size_t)D_MODEL * D_MODEL * 2;
  bf16* WkT  = (bf16*)(ws + o); o += (size_t)D_MODEL * D_MODEL * 2;
  bf16* WvT  = (bf16*)(ws + o); o += (size_t)D_MODEL * D_MODEL * 2;
  bf16* WoT  = (bf16*)(ws + o); o += (size_t)D_MODEL * D_MODEL * 2;
  bf16* W1T  = (bf16*)(ws + o); o += (size_t)D_MODEL * D_FF * 2;
  bf16* W2T  = (bf16*)(ws + o); o += (size_t)D_FF * D_MODEL * 2;
  const size_t qkv_off = o;
  bf16* Qbuf = (bf16*)(ws + o); o += nAct * 2;
  bf16* Kbuf = (bf16*)(ws + o); o += nAct * 2;
  bf16* Vt   = (bf16*)(ws + o); o += nAct * 2;
  bf16* ctx  = (bf16*)(ws + o); o += nAct * 2;
  bf16* hbuf = (bf16*)(ws + qkv_off);                  // aliases Q/K/V/ctx (dead by then)
  float* x1  = (float*)(ws + o); o += nAct * 4;
  float* xf  = (float*)(ws + o); o += nAct * 4;
  bf16*  xb  = (bf16*)(ws + o);  o += nAct * 2;
  float* stats = (float*)(ws + o);                     // 8 x 512 floats

  hipMemsetAsync(stats, 0, 8 * D_MODEL * sizeof(float), stream);

  cvt_bf16  <<<1024, 256, 0, stream>>>(src, srcb, nAct);
  cvt_bf16_T<<<256,  256, 0, stream>>>(W_Q, WqT, D_MODEL, D_MODEL);
  cvt_bf16_T<<<256,  256, 0, stream>>>(W_K, WkT, D_MODEL, D_MODEL);
  cvt_bf16_T<<<256,  256, 0, stream>>>(W_V, WvT, D_MODEL, D_MODEL);
  cvt_bf16_T<<<256,  256, 0, stream>>>(W_O, WoT, D_MODEL, D_MODEL);
  cvt_bf16_T<<<1024, 256, 0, stream>>>(W1, W1T, D_MODEL, D_FF);
  cvt_bf16_T<<<1024, 256, 0, stream>>>(W2, W2T, D_FF, D_MODEL);

  dim3 gP(D_MODEL / 64, ROWS / 128);                   // (8, 128)
  gemm_wmma<EPI_BF16><<<gP, 256, 0, stream>>>(srcb, WqT, nullptr, nullptr, nullptr, Qbuf,
                                              ROWS, D_MODEL, D_MODEL, 0.125f);
  gemm_wmma<EPI_BF16><<<gP, 256, 0, stream>>>(srcb, WkT, nullptr, nullptr, nullptr, Kbuf,
                                              ROWS, D_MODEL, D_MODEL, 1.0f);
  gemm_wmma<EPI_VT><<<gP, 256, 0, stream>>>(srcb, WvT, nullptr, nullptr, nullptr, Vt,
                                            ROWS, D_MODEL, D_MODEL, 1.0f);

  flash_attn<<<dim3(SEQ / 64, BATCH * N_HEADS), 128, 0, stream>>>(Qbuf, Kbuf, Vt, ctx);

  gemm_wmma<EPI_RESF32><<<gP, 256, 0, stream>>>(ctx, WoT, nullptr, src, x1, nullptr,
                                                ROWS, D_MODEL, D_MODEL, 1.0f);

  bn_stats   <<<128, 256, 0, stream>>>(x1, stats, stats + 512);
  bn_finalize<<<1,   512, 0, stream>>>(stats, stats + 512, gamma1, beta1,
                                       stats + 1024, stats + 1536);
  bn_apply   <<<1024, 256, 0, stream>>>(x1, stats + 1024, stats + 1536, xf, xb, nAct);

  dim3 gF1(D_FF / 64, ROWS / 128);                     // (32, 128)
  gemm_wmma<EPI_GELU><<<gF1, 256, 0, stream>>>(xb, W1T, b1, nullptr, nullptr, hbuf,
                                               ROWS, D_FF, D_MODEL, 1.0f);
  gemm_wmma<EPI_RESF32><<<gP, 256, 0, stream>>>(hbuf, W2T, b2, xf, (float*)d_out, nullptr,
                                                ROWS, D_MODEL, D_FF, 1.0f);

  bn_stats   <<<128, 256, 0, stream>>>((float*)d_out, stats + 2048, stats + 2560);
  bn_finalize<<<1,   512, 0, stream>>>(stats + 2048, stats + 2560, gamma2, beta2,
                                       stats + 3072, stats + 3584);
  bn_apply   <<<1024, 256, 0, stream>>>((float*)d_out, stats + 3072, stats + 3584,
                                        (float*)d_out, nullptr, nAct);
}
